// SuperBertSelfAttention_35828617183401
// MI455X (gfx1250) — matrix operand order repaired and checked by reading, not compiled, not run on hardware
//
#include <hip/hip_runtime.h>
#include <hip/hip_bf16.h>

// ---------------------------------------------------------------------------
// CDNA5 (gfx1250) bf16 WMMA attention: QKV GEMM + Toeplitz rel-position panels
// + transposed-score flash attention.  Wave32, V_WMMA_F32_16X16X32_BF16.
// This revision double-buffers operand loads across k-iterations so the
// next step's loads are in flight while the current step's WMMAs execute
// (overlapping live ranges force distinct registers -> partial load waits).
// All prefetch over-reads on the final iteration stay inside d_ws.
// ---------------------------------------------------------------------------

typedef __bf16 v16bf  __attribute__((ext_vector_type(16)));
typedef __bf16 bf16x8 __attribute__((ext_vector_type(8)));
typedef float  v8f    __attribute__((ext_vector_type(8)));
typedef float  f32x4  __attribute__((ext_vector_type(4)));

#define WMMA_BF16(A, B, C) \
  __builtin_amdgcn_wmma_f32_16x16x32_bf16(false, (A), false, (B), (short)0, (C), false, false)

// A-operand (16x32, M rows per lane): lane group g holds K[8g..8g+7], K[16+8g..23+8g].
__device__ __forceinline__ v16bf load_opA(const __bf16* row, int g) {
  bf16x8 lo = *(const bf16x8*)(row + 8 * g);
  bf16x8 hi = *(const bf16x8*)(row + 16 + 8 * g);
  return __builtin_shufflevector(lo, hi, 0,1,2,3,4,5,6,7,8,9,10,11,12,13,14,15);
}

// B-operand (32x16, N col per lane): lane group g holds K[8g..8g+7], K[16+8g..23+8g].
__device__ __forceinline__ v16bf load_opB(const __bf16* row, int g) {
  bf16x8 lo = *(const bf16x8*)(row + 8 * g);
  bf16x8 hi = *(const bf16x8*)(row + 16 + 8 * g);
  return __builtin_shufflevector(lo, hi, 0,1,2,3,4,5,6,7,8,9,10,11,12,13,14,15);
}

// ---------------------------------------------------------------------------
// Kernel 1: f32 -> bf16 conversion (grid-stride)
// ---------------------------------------------------------------------------
__global__ void cvt_f32_bf16(const float* __restrict__ src, __bf16* __restrict__ dst, int n) {
  int stride = gridDim.x * blockDim.x;
  for (int i = blockIdx.x * blockDim.x + threadIdx.x; i < n; i += stride)
    dst[i] = (__bf16)src[i];
}

__global__ void zero_bf16_k(__bf16* __restrict__ dst, int n) {
  int i = blockIdx.x * blockDim.x + threadIdx.x;
  if (i < n) dst[i] = (__bf16)0.0f;
}

// ---------------------------------------------------------------------------
// Kernel 2: QKV projection GEMM.  out[n,m] = sum_k X[m,k] * W[n,k] + bias[n]
// One wave computes a 16(M) x 64(N) tile of one of Q/K/V.
// Q,K -> [B,H,S,D] bf16.  V -> transposed [B,H,D,S] bf16 (packed b128 stores).
// ---------------------------------------------------------------------------
__global__ void qkv_gemm(const __bf16* __restrict__ X,
                         const __bf16* __restrict__ Wq, const __bf16* __restrict__ Wk,
                         const __bf16* __restrict__ Wv,
                         const float* __restrict__ bq, const float* __restrict__ bk,
                         const float* __restrict__ bv,
                         __bf16* __restrict__ Q, __bf16* __restrict__ K,
                         __bf16* __restrict__ Vt) {
  int w    = blockIdx.x * (blockDim.x >> 5) + (threadIdx.x >> 5);
  int lane = threadIdx.x & 31;
  int g = lane >> 4, lm = lane & 15;
  int mt  = w / 48;           // 256 M-tiles of 16 rows (B*S = 4096)
  int rem = w % 48;
  int mat = rem >> 4;         // 0=Q 1=K 2=V
  int n0  = (rem & 15) << 6;  // 16 N-groups of 64

  const __bf16* W    = (mat == 0) ? Wq : ((mat == 1) ? Wk : Wv);
  const float*  bias = (mat == 0) ? bq : ((mat == 1) ? bk : bv);

  const __bf16* xrow = X + (size_t)(mt * 16 + lm) * 1024;
  const __bf16* wrow = W + (size_t)(n0 + lm) * 1024;   // j advances by 16*1024

  // --- software pipeline: preload k-step 0 ---
  v16bf a_cur = load_opA(xrow, g);
  v16bf b_cur[4];
#pragma unroll
  for (int j = 0; j < 4; ++j) b_cur[j] = load_opB(wrow + (size_t)j * 16 * 1024, g);

  v8f acc[4] = {};
  for (int k0 = 0; k0 < 1024; k0 += 32) {
    // prefetch next k-step (final-iteration over-read stays inside d_ws)
    int kn = k0 + 32;
    v16bf a_nxt = load_opA(xrow + kn, g);
    v16bf b_nxt[4];
#pragma unroll
    for (int j = 0; j < 4; ++j)
      b_nxt[j] = load_opB(wrow + (size_t)j * 16 * 1024 + kn, g);

    // compute current k-step while next loads are in flight
#pragma unroll
    for (int j = 0; j < 4; ++j)
      acc[j] = WMMA_BF16(a_cur, b_cur[j], acc[j]);

    a_cur = a_nxt;
#pragma unroll
    for (int j = 0; j < 4; ++j) b_cur[j] = b_nxt[j];
  }

  int b_idx  = (mt * 16) >> 10;   // batch (16-row tiles never cross batch)
  int s_base = (mt * 16) & 1023;
#pragma unroll
  for (int j = 0; j < 4; ++j) {
    int n = n0 + j * 16 + lm;     // this lane's output column
    float bb = bias[n];
    int h = n >> 6, d = n & 63;
    if (mat < 2) {
      __bf16* out = (mat == 0) ? Q : K;
#pragma unroll
      for (int v = 0; v < 8; ++v) {
        int s = s_base + v + 8 * g;               // C layout: M = v + 8g
        out[((size_t)(b_idx * 16 + h) * 1024 + s) * 64 + d] = (__bf16)(acc[j][v] + bb);
      }
    } else {
      bf16x8 pk;
#pragma unroll
      for (int v = 0; v < 8; ++v) pk[v] = (__bf16)(acc[j][v] + bb);
      int s = s_base + 8 * g;                     // 8 consecutive s -> one b128
      *(bf16x8*)(Vt + ((size_t)(b_idx * 16 + h) * 64 + d) * 1024 + s) = pk;
    }
  }
}

// ---------------------------------------------------------------------------
// Kernel 3: relative-position panels.
// panel[j'][m] = X[bh, t*16+m] . E[jbase + j'],  j' in [0,1040),
// jbase = t*16 (Q) or 1008 - t*16 (K).  E padded to 2048 rows (row 2047 = 0).
// Stored transposed bf16: P[(jc*16+jlane)*16 + m].
// ---------------------------------------------------------------------------
__global__ void rel_panels(const __bf16* __restrict__ Q, const __bf16* __restrict__ K,
                           const __bf16* __restrict__ E,
                           __bf16* __restrict__ QEp, __bf16* __restrict__ KEp) {
  int w    = blockIdx.x * (blockDim.x >> 5) + (threadIdx.x >> 5);  // [0, 8192)
  int lane = threadIdx.x & 31;
  int g = lane >> 4, lm = lane & 15;
  int which = w >> 12;            // 0 = Q panels, 1 = K panels
  int bh    = (w >> 6) & 63;
  int t     = w & 63;

  const __bf16* X = (which == 0) ? Q : K;
  __bf16* P = ((which == 0) ? QEp : KEp) + ((size_t)bh * 64 + t) * 16640;
  int jbase = (which == 0) ? (t * 16) : (1008 - t * 16);

  const __bf16* xrow = X + ((size_t)bh * 1024 + t * 16 + lm) * 64;
  v16bf a0 = load_opA(xrow, g);        // d 0..31
  v16bf a1 = load_opA(xrow + 32, g);   // d 32..63

  const __bf16* erow = E + (size_t)(jbase + lm) * 64;
  v16bf b0_cur = load_opB(erow, g);
  v16bf b1_cur = load_opB(erow + 32, g);

  for (int jc = 0; jc < 65; ++jc) {
    // prefetch next column tile (final over-read of <=16 E rows stays in d_ws)
    const __bf16* erow_n = erow + 16 * 64;
    v16bf b0_nxt = load_opB(erow_n, g);
    v16bf b1_nxt = load_opB(erow_n + 32, g);

    v8f acc = {};
    acc = WMMA_BF16(a0, b0_cur, acc);
    acc = WMMA_BF16(a1, b1_cur, acc);
    bf16x8 pk;
#pragma unroll
    for (int v = 0; v < 8; ++v) pk[v] = (__bf16)acc[v];
    *(bf16x8*)(P + (size_t)(jc * 16 + lm) * 16 + 8 * g) = pk;   // m = v + 8g

    erow = erow_n;
    b0_cur = b0_nxt;
    b1_cur = b1_nxt;
  }
}

// ---------------------------------------------------------------------------
// Kernel 4: flash attention, transposed scores.  One wave per (b,h,q-tile).
// S^T tile layout: lane lm = q column, vgpr r -> k row = k0 + 8g + r.
// The two S^T tiles (k0, k0+16) pack exactly into the PV A-operand per lane.
// Logits ~ N(0,1) (bounded) -> plain exp, single normalize at the end.
// K/V operands are double-buffered across the k loop.
// ---------------------------------------------------------------------------
__global__ void flash_attn(const __bf16* __restrict__ Q, const __bf16* __restrict__ K,
                           const __bf16* __restrict__ Vt,
                           const __bf16* __restrict__ QEp, const __bf16* __restrict__ KEp,
                           const float* __restrict__ mask, float* __restrict__ out) {
  int w    = blockIdx.x * (blockDim.x >> 5) + (threadIdx.x >> 5);  // [0, 4096)
  int lane = threadIdx.x & 31;
  int g = lane >> 4, lm = lane & 15;
  int bh = w >> 6, qt = w & 63;
  int b = bh >> 4, h = bh & 15;
  int q0 = qt * 16;

  // Q as B-operand (lane = q column), contiguous d chunks
  const __bf16* qrow = Q + ((size_t)bh * 1024 + q0 + lm) * 64;
  v16bf qb0 = load_opB(qrow, g);        // d 0..31
  v16bf qb1 = load_opB(qrow + 32, g);   // d 32..63

  const __bf16* Qpan  = QEp + ((size_t)bh * 64 + qt) * 16640;
  const __bf16* Kpan  = KEp + (size_t)bh * 64 * 16640;
  const float*  mrow  = mask + (size_t)b * 1024;
  const __bf16* kbase_p = K + ((size_t)bh * 1024 + lm) * 64;     // + k*64
  const __bf16* vbase_p = Vt + (size_t)bh * 64 * 1024;           // + d*1024 + k

  // --- software pipeline: preload k-step 0 ---
  v16bf kc[4], vc[4];
  kc[0] = load_opA(kbase_p, g);
  kc[1] = load_opA(kbase_p + 32, g);
  kc[2] = load_opA(kbase_p + 16 * 64, g);
  kc[3] = load_opA(kbase_p + 16 * 64 + 32, g);
#pragma unroll
  for (int dt = 0; dt < 4; ++dt)
    vc[dt] = load_opB(vbase_p + (size_t)(dt * 16 + lm) * 1024, g);

  v8f ctx[4] = {};
  float lsum = 0.0f;

  for (int k0 = 0; k0 < 1024; k0 += 32) {
    // prefetch next k-step (final over-read stays inside d_ws)
    const __bf16* krn = kbase_p + (size_t)(k0 + 32) * 64;
    v16bf kn[4], vn[4];
    kn[0] = load_opA(krn, g);
    kn[1] = load_opA(krn + 32, g);
    kn[2] = load_opA(krn + 16 * 64, g);
    kn[3] = load_opA(krn + 16 * 64 + 32, g);
    const __bf16* vrn = vbase_p + k0 + 32;
#pragma unroll
    for (int dt = 0; dt < 4; ++dt)
      vn[dt] = load_opB(vrn + (size_t)(dt * 16 + lm) * 1024, g);

    // vectorized mask loads: contiguous f32, 16B-aligned (kbase multiple of 8)
    int kb0 = k0 + 8 * g;
    f32x4 mk0a = *(const f32x4*)(mrow + kb0);
    f32x4 mk0b = *(const f32x4*)(mrow + kb0 + 4);
    f32x4 mk1a = *(const f32x4*)(mrow + kb0 + 16);
    f32x4 mk1b = *(const f32x4*)(mrow + kb0 + 20);

    v8f st0 = {}, st1 = {};
    st0 = WMMA_BF16(kc[0], qb0, st0);
    st0 = WMMA_BF16(kc[1], qb1, st0);
    st1 = WMMA_BF16(kc[2], qb0, st1);
    st1 = WMMA_BF16(kc[3], qb1, st1);

    v16bf pa;
#pragma unroll
    for (int r = 0; r < 8; ++r) {
      int ka = kb0 + r;          // st0 absolute k
      int kb = ka + 16;          // st1 absolute k
      // rel_q gather: panel row = lm, j' = q - k + 1023 - q0 = lm + 1023 - k
      float rq0 = (float)Qpan[(size_t)(lm + 1023 - ka) * 16 + lm];
      float rq1 = (float)Qpan[(size_t)(lm + 1023 - kb) * 16 + lm];
      // rel_k gather: k-tile panel, row = k&15, j' = q - (k&15) + 15
      float rk0 = (float)Kpan[(size_t)(ka >> 4) * 16640 +
                              (size_t)(q0 + lm - (ka & 15) + 15) * 16 + (ka & 15)];
      float rk1 = (float)Kpan[(size_t)(kb >> 4) * 16640 +
                              (size_t)(q0 + lm - (kb & 15) + 15) * 16 + (kb & 15)];
      float mk0 = (r < 4) ? mk0a[r & 3] : mk0b[r & 3];
      float mk1 = (r < 4) ? mk1a[r & 3] : mk1b[r & 3];
      float s0 = (st0[r] + rq0 + rk0) * 0.125f + mk0;
      float s1 = (st1[r] + rq1 + rk1) * 0.125f + mk1;
      float p0 = __expf(s0);
      float p1 = __expf(s1);
      lsum += p0 + p1;
      pa[r]     = (__bf16)p0;    // A halves: st0 -> K[8g..], st1 -> K[16+8g..]
      pa[r + 8] = (__bf16)p1;
    }

#pragma unroll
    for (int dt = 0; dt < 4; ++dt)
      ctx[dt] = WMMA_BF16(pa, vc[dt], ctx[dt]);

    // rotate pipeline buffers
#pragma unroll
    for (int dt = 0; dt < 4; ++dt) { kc[dt] = kn[dt]; vc[dt] = vn[dt]; }
  }

  // combine the two k-halves of each q row, normalize
  lsum += __shfl_xor(lsum, 16, 32);
  float linv = 1.0f / lsum;
  float linv_v[8];
#pragma unroll
  for (int v = 0; v < 8; ++v)
    linv_v[v] = __shfl(linv, v + 8 * g, 32);   // broadcast 1/l for row q = v + 8g

  // store ctx -> out [B, S, H*D] f32 (coalesced across lanes in d)
#pragma unroll
  for (int dt = 0; dt < 4; ++dt) {
    int d = dt * 16 + lm;
#pragma unroll
    for (int v = 0; v < 8; ++v) {
      int q = q0 + v + 8 * g;
      out[((size_t)b * 1024 + q) * 1024 + h * 64 + d] = ctx[dt][v] * linv_v[v];
    }
  }
}

// ---------------------------------------------------------------------------
// Host launcher.  Workspace layout (bytes):
//   hb 0 | wq 8388608 | wk 10485760 | wv 12582912 | E 14680064 (2048x64)
//   Qb 14942208 | Kb 23330816 | Vt 31719424 | QEp 40108032 | KEp 176422912
// ---------------------------------------------------------------------------
extern "C" void kernel_launch(void* const* d_in, const int* in_sizes, int n_in,
                              void* d_out, int out_size, void* d_ws, size_t ws_size,
                              hipStream_t stream) {
  const float* hidden = (const float*)d_in[0];
  const float* mask   = (const float*)d_in[1];
  const float* Wq     = (const float*)d_in[2];
  const float* bq     = (const float*)d_in[3];
  const float* Wk     = (const float*)d_in[4];
  const float* bk     = (const float*)d_in[5];
  const float* Wv     = (const float*)d_in[6];
  const float* bv     = (const float*)d_in[7];
  const float* dist   = (const float*)d_in[8];

  char* ws = (char*)d_ws;
  __bf16* hb  = (__bf16*)(ws + 0);
  __bf16* wqb = (__bf16*)(ws + 8388608);
  __bf16* wkb = (__bf16*)(ws + 10485760);
  __bf16* wvb = (__bf16*)(ws + 12582912);
  __bf16* eb  = (__bf16*)(ws + 14680064);
  __bf16* qb  = (__bf16*)(ws + 14942208);
  __bf16* kb  = (__bf16*)(ws + 23330816);
  __bf16* vt  = (__bf16*)(ws + 31719424);
  __bf16* qep = (__bf16*)(ws + 40108032);
  __bf16* kep = (__bf16*)(ws + 176422912);

  cvt_f32_bf16<<<1024, 256, 0, stream>>>(hidden, hb, 4096 * 1024);
  cvt_f32_bf16<<<512,  256, 0, stream>>>(Wq, wqb, 1024 * 1024);
  cvt_f32_bf16<<<512,  256, 0, stream>>>(Wk, wkb, 1024 * 1024);
  cvt_f32_bf16<<<512,  256, 0, stream>>>(Wv, wvb, 1024 * 1024);
  cvt_f32_bf16<<<256,  256, 0, stream>>>(dist, eb, 2047 * 64);
  zero_bf16_k<<<1, 64, 0, stream>>>(eb + 2047 * 64, 64);   // pad row

  qkv_gemm<<<1536, 256, 0, stream>>>(hb, wqb, wkb, wvb, bq, bk, bv, qb, kb, vt);
  rel_panels<<<1024, 256, 0, stream>>>(qb, kb, eb, qep, kep);
  flash_attn<<<512, 256, 0, stream>>>(qb, kb, vt, qep, kep, mask, (float*)d_out);
}
